// GNN_39393440039358
// MI455X (gfx1250) — compile-verified, matrix operand order checked by reading, not run on hardware
//
#include <hip/hip_runtime.h>
#include <hip/hip_bf16.h>

// ---------------------------------------------------------------------------
// GNN forward (RGCN -> TransformerConv -> SAGE) for MI455X / gfx1250.
// Dense GEMMs use v_wmma_f32_16x16x32_f16 with register-resident B fragments
// (one wave sweeps a group of 16-row tiles, reusing B across the group).
// Sparse segment ops use global float atomics (L2-resident tables).
// NOTE: all GEMMs here run with N a multiple of 16 (N=50000=16*3125), so the
// row-store path is unguarded (fully coalesced b32 stores, no exec churn).
// ---------------------------------------------------------------------------

#define NN   50000
#define EE   800000
#define RR   8
#define GG   64
#define H1D  64
#define H2D  32
#define HEADS 4
#define HOUT1 576          // 8*64 relations + 64 root
#define HOUT2 512          // q|k|v|skip, 128 each
#define HOUT3 32

typedef __attribute__((ext_vector_type(16))) _Float16 v16h;
typedef __attribute__((ext_vector_type(8)))  _Float16 v8h;
typedef __attribute__((ext_vector_type(8)))  float    v8f;

// ------------------------------ utility ------------------------------------

__global__ void fill_f32_k(float* p, float v, int n) {
  int i = blockIdx.x * blockDim.x + threadIdx.x;
  if (i < n) p[i] = v;
}
__global__ void fill_u32_k(unsigned* p, unsigned v, int n) {
  int i = blockIdx.x * blockDim.x + threadIdx.x;
  if (i < n) p[i] = v;
}
__global__ void cast_half_k(const float* __restrict__ x, _Float16* __restrict__ y, int n) {
  int i = blockIdx.x * blockDim.x + threadIdx.x;
  if (i < n) y[i] = (_Float16)x[i];
}

__device__ __forceinline__ unsigned fkey(float x) {
  unsigned b = __float_as_uint(x);
  return (b & 0x80000000u) ? ~b : (b | 0x80000000u);
}
__device__ __forceinline__ float funkey(unsigned k) {
  unsigned b = (k & 0x80000000u) ? (k & 0x7FFFFFFFu) : ~k;
  return __uint_as_float(b);
}

// --------------------------- weight packing --------------------------------
// Packed B layout for WMMA: [tileH][K/32][lane 0..31][16 halfs contiguous].
// Fragment element e of lane l maps to B[k][n] with
//   n = tile*16 + (l&15),  k = c*32 + (e>>3)*16 + (l>>4)*8 + (e&7).

__global__ void pack_w1_k(const float* __restrict__ rgcn_w,   // [8,64,64]
                          const float* __restrict__ rgcn_root,// [64,64]
                          _Float16* __restrict__ Wp) {        // 36 tiles * 2 chunks
  int idx = blockIdx.x * blockDim.x + threadIdx.x;
  if (idx >= 36 * 2 * 32 * 16) return;
  int e = idx & 15, lane = (idx >> 4) & 31, c = (idx >> 9) & 1, tile = idx >> 10;
  int n = tile * 16 + (lane & 15);
  int k = c * 32 + ((e >> 3) << 4) + ((lane >> 4) << 3) + (e & 7);
  float v;
  if (n < 512) { int r = n >> 6, h = n & 63; v = rgcn_w[(r * 64 + k) * 64 + h]; }
  else         { v = rgcn_root[k * 64 + (n - 512)]; }
  Wp[idx] = (_Float16)v;
}

__global__ void pack_w2_k(const float* __restrict__ wq, const float* __restrict__ wk,
                          const float* __restrict__ wv, const float* __restrict__ ws,
                          _Float16* __restrict__ Wp) {        // 32 tiles * 2 chunks
  int idx = blockIdx.x * blockDim.x + threadIdx.x;
  if (idx >= 32 * 2 * 32 * 16) return;
  int e = idx & 15, lane = (idx >> 4) & 31, c = (idx >> 9) & 1, tile = idx >> 10;
  int n = tile * 16 + (lane & 15);
  int k = c * 32 + ((e >> 3) << 4) + ((lane >> 4) << 3) + (e & 7);
  const float* W = (n < 128) ? wq : (n < 256) ? wk : (n < 384) ? wv : ws;
  Wp[idx] = (_Float16)W[k * 128 + (n & 127)];
}

__global__ void pack_w3_k(const float* __restrict__ wl, const float* __restrict__ wr,
                          _Float16* __restrict__ Wp) {        // 2 tiles * 8 chunks, K=256
  int idx = blockIdx.x * blockDim.x + threadIdx.x;
  if (idx >= 2 * 8 * 32 * 16) return;
  int e = idx & 15, lane = (idx >> 4) & 31, c = (idx >> 9) & 7, tile = idx >> 12;
  int n = tile * 16 + (lane & 15);
  int k = c * 32 + ((e >> 3) << 4) + ((lane >> 4) << 3) + (e & 7);
  float v = (k < 128) ? wl[k * 32 + n] : wr[(k - 128) * 32 + n];
  Wp[idx] = (_Float16)v;
}

__global__ void pack_bias2_k(const float* bq, const float* bk, const float* bv,
                             const float* bs, float* out) {
  int i = blockIdx.x * blockDim.x + threadIdx.x;
  if (i >= 512) return;
  const float* B = (i < 128) ? bq : (i < 256) ? bk : (i < 384) ? bv : bs;
  out[i] = B[i & 127];
}

// ------------------------------ WMMA GEMM ----------------------------------
// One wave owns one 16-wide output column tile (tileH). It hoists all KC
// B-fragments into registers once, then sweeps GROUP consecutive 16-row tiles,
// so each v_wmma needs only two 16B A-loads (B traffic fully amortized).
// Requires N % 16 == 0 (true for all call sites): row stores are unguarded.

template <int KC, int GROUP>
__global__ void __launch_bounds__(32)
wmma_gemm_k(const _Float16* __restrict__ A, const _Float16* __restrict__ Bp,
            const float* __restrict__ bias, float* __restrict__ Y,
            int N, int Hout) {
  const int K = KC * 32;
  int tileH = blockIdx.y;
  int lane = threadIdx.x;
  int m = lane & 15, hl = lane >> 4;

  // hoist B fragments (per-lane contiguous 32B each) into registers
  const _Float16* bp = Bp + (((size_t)tileH * KC) * 32 + lane) * 16;
  v16h Breg[KC];
#pragma unroll
  for (int c = 0; c < KC; ++c) Breg[c] = *(const v16h*)(bp + (size_t)c * 512);

  int col = tileH * 16 + m;
  float bv = bias ? bias[col] : 0.0f;

  int tN0 = blockIdx.x * GROUP;
#pragma unroll
  for (int g = 0; g < GROUP; ++g) {
    int tileN = tN0 + g;
    if (tileN * 16 >= N) break;                 // uniform across the wave
    const _Float16* arow = A + (size_t)(tileN * 16 + m) * K;

    v8f acc = {};
#pragma unroll
    for (int c = 0; c < KC; ++c) {
      v8h alo = *(const v8h*)(arow + c * 32 + hl * 8);
      v8h ahi = *(const v8h*)(arow + c * 32 + 16 + hl * 8);
      v16h a;
#pragma unroll
      for (int i = 0; i < 8; ++i) { a[i] = alo[i]; a[i + 8] = ahi[i]; }
      acc = __builtin_amdgcn_wmma_f32_16x16x32_f16(
          /*neg_a=*/false, a, /*neg_b=*/false, Breg[c],
          /*c_mod=*/(short)0, acc, /*reuse_a=*/false, /*reuse_b=*/false);
    }
    // N % 16 == 0 and tileN*16 < N  =>  all 16 rows in range: no guard.
    float* yrow = Y + (size_t)(tileN * 16 + hl * 8) * Hout + col;
#pragma unroll
    for (int j = 0; j < 8; ++j) {
      yrow[(size_t)j * Hout] = acc[j] + bv;
    }
  }
}

// ------------------------------- layer 1 -----------------------------------

__global__ void rgcn_count_k(const int* __restrict__ dst, const int* __restrict__ et,
                             float* __restrict__ cnt, int E) {
  int e = blockIdx.x * blockDim.x + threadIdx.x;
  if (e < E) atomicAdd(&cnt[dst[e] * RR + et[e]], 1.0f);
}

__global__ void rgcn_scatter_k(const float* __restrict__ XR, const int* __restrict__ src,
                               const int* __restrict__ dst, const int* __restrict__ et,
                               const float* __restrict__ cnt, float* __restrict__ agg) {
  int e = blockIdx.x, f = threadIdx.x;   // 64 threads
  int s = src[e], d = dst[e], r = et[e];
  float inv = 1.0f / fmaxf(cnt[d * RR + r], 1.0f);
  atomicAdd(&agg[(size_t)d * H1D + f], XR[(size_t)s * HOUT1 + r * H1D + f] * inv);
}

__global__ void combine1_k(const float* __restrict__ agg, const float* __restrict__ XR,
                           const float* __restrict__ bias, float* __restrict__ H, int N) {
  int t = blockIdx.x * blockDim.x + threadIdx.x;
  if (t >= N * H1D) return;
  int n = t >> 6, f = t & 63;
  H[t] = agg[t] + XR[(size_t)n * HOUT1 + 512 + f] + bias[f];
}

// ------------------------------ BatchNorm ----------------------------------

__global__ void bn_stats_k(const float* __restrict__ X, int N, int C,
                           float* __restrict__ mean, float* __restrict__ var) {
  int c = blockIdx.x;
  float s = 0.f, s2 = 0.f;
  for (int n = threadIdx.x; n < N; n += blockDim.x) {
    float v = X[(size_t)n * C + c];
    s += v; s2 += v * v;
  }
  __shared__ float sh[256], sh2[256];
  sh[threadIdx.x] = s; sh2[threadIdx.x] = s2;
  __syncthreads();
  for (int o = 128; o > 0; o >>= 1) {
    if (threadIdx.x < o) { sh[threadIdx.x] += sh[threadIdx.x + o]; sh2[threadIdx.x] += sh2[threadIdx.x + o]; }
    __syncthreads();
  }
  if (threadIdx.x == 0) {
    float m = sh[0] / (float)N;
    mean[c] = m;
    var[c] = sh2[0] / (float)N - m * m;
  }
}

__global__ void bn_apply_res_k(const float* __restrict__ H, const float* __restrict__ mean,
                               const float* __restrict__ var, const float* __restrict__ g,
                               const float* __restrict__ b, const float* __restrict__ res,
                               int resC, float* __restrict__ out, _Float16* __restrict__ outh,
                               int N, int C) {
  int t = blockIdx.x * blockDim.x + threadIdx.x;
  if (t >= N * C) return;
  int n = t / C, c = t % C;
  float x = H[t];
  float y = g[c] * (x - mean[c]) * rsqrtf(var[c] + 1e-5f) + b[c];
  y = (y > 0.f) ? y : 0.01f * y;                 // leaky_relu
  if (c < resC) y += res[(size_t)n * resC + c];  // pad-or-truncate residual
  out[t] = y;
  if (outh) outh[t] = (_Float16)y;
}

// ------------------------------- layer 2 -----------------------------------

__global__ void attn_logits_k(const float* __restrict__ QKVS, const int* __restrict__ src,
                              const int* __restrict__ dst, float* __restrict__ logits,
                              unsigned* __restrict__ mxkey, int E) {
  int t = blockIdx.x * blockDim.x + threadIdx.x;
  if (t >= E * HEADS) return;
  int e = t >> 2, h = t & 3;
  int s = src[e], d = dst[e];
  const float* q = QKVS + (size_t)d * HOUT2 + h * H2D;         // q cols 0..127
  const float* k = QKVS + (size_t)s * HOUT2 + 128 + h * H2D;   // k cols 128..255
  float acc = 0.f;
#pragma unroll
  for (int i = 0; i < H2D; ++i) acc += q[i] * k[i];
  acc *= 0.17677669529663687f;                                 // 1/sqrt(32)
  logits[t] = acc;
  atomicMax(&mxkey[d * HEADS + h], fkey(acc));
}

__global__ void attn_ex_k(const int* __restrict__ dst, const unsigned* __restrict__ mxkey,
                          float* __restrict__ logits, float* __restrict__ den, int E) {
  int t = blockIdx.x * blockDim.x + threadIdx.x;
  if (t >= E * HEADS) return;
  int e = t >> 2, h = t & 3;
  int d = dst[e];
  float ex = __expf(logits[t] - funkey(mxkey[d * HEADS + h]));
  logits[t] = ex;                                // in-place: logits -> ex
  atomicAdd(&den[d * HEADS + h], ex);
}

__global__ void attn_scatter_k(const float* __restrict__ QKVS, const float* __restrict__ ex,
                               const float* __restrict__ den, const int* __restrict__ src,
                               const int* __restrict__ dst, float* __restrict__ outb) {
  int e = blockIdx.x, f = threadIdx.x;   // 128 threads
  int h = f >> 5;
  int s = src[e], d = dst[e];
  float w = ex[e * HEADS + h] / fmaxf(den[d * HEADS + h], 1e-16f);
  atomicAdd(&outb[(size_t)d * 128 + f], w * QKVS[(size_t)s * HOUT2 + 256 + f]); // v cols 256..383
}

__global__ void combine2_k(const float* __restrict__ attnout, const float* __restrict__ QKVS,
                           float* __restrict__ H, int N) {
  int t = blockIdx.x * blockDim.x + threadIdx.x;
  if (t >= N * 128) return;
  int n = t >> 7, j = t & 127;
  H[t] = attnout[t] + QKVS[(size_t)n * HOUT2 + 384 + j];       // skip cols 384..511
}

// ------------------------------- layer 3 -----------------------------------

__global__ void sage_scatter_k(const float* __restrict__ X2, const int* __restrict__ src,
                               const int* __restrict__ dst, float* __restrict__ nsum,
                               float* __restrict__ ncnt) {
  int e = blockIdx.x, f = threadIdx.x;   // 128 threads
  int s = src[e], d = dst[e];
  atomicAdd(&nsum[(size_t)d * 128 + f], X2[(size_t)s * 128 + f]);
  if (f == 0) atomicAdd(&ncnt[d], 1.0f);
}

__global__ void build_xcat_k(const float* __restrict__ nsum, const float* __restrict__ ncnt,
                             const float* __restrict__ X2, _Float16* __restrict__ Xcat, int N) {
  int t = blockIdx.x * blockDim.x + threadIdx.x;
  if (t >= N * 256) return;
  int n = t >> 8, j = t & 255;
  float v = (j < 128) ? nsum[(size_t)n * 128 + j] / fmaxf(ncnt[n], 1.0f)
                      : X2[(size_t)n * 128 + (j - 128)];
  Xcat[t] = (_Float16)v;
}

// ------------------------------ launcher -----------------------------------

static inline int cdiv(long long a, long long b) { return (int)((a + b - 1) / b); }

extern "C" void kernel_launch(void* const* d_in, const int* in_sizes, int n_in,
                              void* d_out, int out_size, void* d_ws, size_t ws_size,
                              hipStream_t stream) {
  const float* node_features = (const float*)d_in[0];
  const int*   edge_index    = (const int*)d_in[2];
  const int*   edge_type     = (const int*)d_in[3];
  const float* rgcn_w        = (const float*)d_in[4];
  const float* rgcn_root     = (const float*)d_in[5];
  const float* rgcn_bias     = (const float*)d_in[6];
  const float* bn1_g = (const float*)d_in[7],  *bn1_b = (const float*)d_in[8];
  const float* wq = (const float*)d_in[9],  *bq = (const float*)d_in[10];
  const float* wk = (const float*)d_in[11], *bk = (const float*)d_in[12];
  const float* wv = (const float*)d_in[13], *bv = (const float*)d_in[14];
  const float* wsk = (const float*)d_in[15], *bsk = (const float*)d_in[16];
  const float* bn2_g = (const float*)d_in[17], *bn2_b = (const float*)d_in[18];
  const float* sage_wl = (const float*)d_in[19], *sage_bl = (const float*)d_in[20];
  const float* sage_wr = (const float*)d_in[21];
  const float* bn3_g = (const float*)d_in[22], *bn3_b = (const float*)d_in[23];

  const int* src = edge_index;          // edge_index[0,:]
  const int* dst = edge_index + EE;     // edge_index[1,:]

  // ---- workspace bump allocator (256B aligned) ----
  char* base = (char*)d_ws;
  size_t off = 0;
  auto alloc = [&](size_t bytes) -> void* {
    void* p = base + off;
    off += (bytes + 255) & ~(size_t)255;
    return p;
  };
  _Float16* Xh    = (_Float16*)alloc((size_t)NN * GG * 2);
  _Float16* Wp1   = (_Float16*)alloc(36 * 2 * 32 * 16 * 2);
  _Float16* Wp2   = (_Float16*)alloc(32 * 2 * 32 * 16 * 2);
  _Float16* Wp3   = (_Float16*)alloc(2 * 8 * 32 * 16 * 2);
  float*    bias2 = (float*)alloc(512 * 4);
  float*    XR    = (float*)alloc((size_t)NN * HOUT1 * 4);   // later reused as QKVS
  float*    QKVS  = XR;                                      // HOUT2*4 <= HOUT1*4
  float*    cnt   = (float*)alloc((size_t)NN * RR * 4);
  float*    agg   = (float*)alloc((size_t)NN * H1D * 4);
  float*    Hbuf  = (float*)alloc((size_t)NN * 128 * 4);     // h1 / h2 / h3
  float*    mean  = (float*)alloc(128 * 4);
  float*    var   = (float*)alloc(128 * 4);
  float*    x1    = (float*)alloc((size_t)NN * H1D * 4);
  _Float16* X1h   = (_Float16*)alloc((size_t)NN * H1D * 2);
  float*    logits= (float*)alloc((size_t)EE * HEADS * 4);
  unsigned* mxkey = (unsigned*)alloc((size_t)NN * HEADS * 4);
  float*    den   = (float*)alloc((size_t)NN * HEADS * 4);
  float*    attno = (float*)alloc((size_t)NN * 128 * 4);
  float*    x2    = (float*)alloc((size_t)NN * 128 * 4);
  _Float16* Xcat  = (_Float16*)alloc((size_t)NN * 256 * 2);
  float*    nsum  = (float*)alloc((size_t)NN * 128 * 4);
  float*    ncnt  = (float*)alloc((size_t)NN * 4);
  (void)ws_size; (void)n_in; (void)in_sizes; (void)out_size;

  const int TB = 256;
  const int GROUP = 5;                  // 3125 N-tiles = 5 * 625 (exact)
  const int tilesN = cdiv(NN, 16);      // 3125
  const int gx = cdiv(tilesN, GROUP);   // 625

  // ---- zero accumulators (fresh every call; graph-replay safe) ----
  fill_f32_k<<<cdiv((size_t)NN * RR, TB), TB, 0, stream>>>(cnt, 0.f, NN * RR);
  fill_f32_k<<<cdiv((size_t)NN * H1D, TB), TB, 0, stream>>>(agg, 0.f, NN * H1D);
  fill_u32_k<<<cdiv((size_t)NN * HEADS, TB), TB, 0, stream>>>(mxkey, 0u, NN * HEADS);
  fill_f32_k<<<cdiv((size_t)NN * HEADS, TB), TB, 0, stream>>>(den, 0.f, NN * HEADS);
  fill_f32_k<<<cdiv((size_t)NN * 128, TB), TB, 0, stream>>>(attno, 0.f, NN * 128);
  fill_f32_k<<<cdiv((size_t)NN * 128, TB), TB, 0, stream>>>(nsum, 0.f, NN * 128);
  fill_f32_k<<<cdiv(NN, TB), TB, 0, stream>>>(ncnt, 0.f, NN);

  // ---- prep: cast activations, pack weights ----
  cast_half_k<<<cdiv((size_t)NN * GG, TB), TB, 0, stream>>>(node_features, Xh, NN * GG);
  pack_w1_k<<<cdiv(36 * 2 * 32 * 16, TB), TB, 0, stream>>>(rgcn_w, rgcn_root, Wp1);
  pack_w2_k<<<cdiv(32 * 2 * 32 * 16, TB), TB, 0, stream>>>(wq, wk, wv, wsk, Wp2);
  pack_w3_k<<<cdiv(2 * 8 * 32 * 16, TB), TB, 0, stream>>>(sage_wl, sage_wr, Wp3);
  pack_bias2_k<<<2, TB, 0, stream>>>(bq, bk, bv, bsk, bias2);

  // ---- layer 1: RGCN ----
  wmma_gemm_k<2, GROUP><<<dim3(gx, HOUT1 / 16), 32, 0, stream>>>(Xh, Wp1, nullptr, XR, NN, HOUT1);
  rgcn_count_k<<<cdiv(EE, TB), TB, 0, stream>>>(dst, edge_type, cnt, EE);
  rgcn_scatter_k<<<EE, H1D, 0, stream>>>(XR, src, dst, edge_type, cnt, agg);
  combine1_k<<<cdiv((size_t)NN * H1D, TB), TB, 0, stream>>>(agg, XR, rgcn_bias, Hbuf, NN);
  bn_stats_k<<<H1D, 256, 0, stream>>>(Hbuf, NN, H1D, mean, var);
  bn_apply_res_k<<<cdiv((size_t)NN * H1D, TB), TB, 0, stream>>>(
      Hbuf, mean, var, bn1_g, bn1_b, node_features, GG, x1, X1h, NN, H1D);

  // ---- layer 2: TransformerConv (XR region reused as QKVS) ----
  wmma_gemm_k<2, GROUP><<<dim3(gx, HOUT2 / 16), 32, 0, stream>>>(X1h, Wp2, bias2, QKVS, NN, HOUT2);
  attn_logits_k<<<cdiv((size_t)EE * HEADS, TB), TB, 0, stream>>>(QKVS, src, dst, logits, mxkey, EE);
  attn_ex_k<<<cdiv((size_t)EE * HEADS, TB), TB, 0, stream>>>(dst, mxkey, logits, den, EE);
  attn_scatter_k<<<EE, 128, 0, stream>>>(QKVS, logits, den, src, dst, attno);
  combine2_k<<<cdiv((size_t)NN * 128, TB), TB, 0, stream>>>(attno, QKVS, Hbuf, NN);
  bn_stats_k<<<128, 256, 0, stream>>>(Hbuf, NN, 128, mean, var);
  bn_apply_res_k<<<cdiv((size_t)NN * 128, TB), TB, 0, stream>>>(
      Hbuf, mean, var, bn2_g, bn2_b, x1, H1D, x2, nullptr, NN, 128);

  // ---- layer 3: SAGE ----
  sage_scatter_k<<<EE, 128, 0, stream>>>(x2, src, dst, nsum, ncnt);
  build_xcat_k<<<cdiv((size_t)NN * 256, TB), TB, 0, stream>>>(nsum, ncnt, x2, Xcat, NN);
  wmma_gemm_k<8, GROUP><<<dim3(gx, HOUT3 / 16), 32, 0, stream>>>(Xcat, Wp3, sage_bl, Hbuf, NN, HOUT3);
  bn_stats_k<<<HOUT3, 256, 0, stream>>>(Hbuf, NN, HOUT3, mean, var);
  bn_apply_res_k<<<cdiv((size_t)NN * HOUT3, TB), TB, 0, stream>>>(
      Hbuf, mean, var, bn3_g, bn3_b, x2, 128, (float*)d_out, nullptr, NN, HOUT3);
}
